// MobileMQA_42752104465005
// MI455X (gfx1250) — matrix operand order repaired and checked by reading, not compiled, not run on hardware
//
#include <hip/hip_runtime.h>
#include <hip/hip_bf16.h>

typedef __bf16 bf16;
typedef __attribute__((ext_vector_type(16))) __bf16 v16bf;
typedef __attribute__((ext_vector_type(8)))  __bf16 v8bf;
typedef __attribute__((ext_vector_type(8)))  float  v8f;

#define DIMC   512
#define NHEADS 8
#define HD     64
#define BATCH  8
#define NQ     1024     // 32*32
#define NKPOS  256      // 16*16
#define MKV    2048     // NKPOS * NHEADS

__device__ __forceinline__ int lane_() { return threadIdx.x & 31; }

__device__ __forceinline__ v8f wmma_bf16(v16bf a, v16bf b, v8f c) {
    // D = A(16x32 bf16) * B(32x16 bf16) + C(16x16 f32)
    return __builtin_amdgcn_wmma_f32_16x16x32_bf16(
        false, a, false, b, (short)0, c, false, false);
}

// ---- A fragment (16x32, bf16), row-major source ----
// lane<16: row=lane,   K elements {k0..k0+15}
// lane>=16: row=lane&15, K elements {k0+16..k0+31}
__device__ __forceinline__ v16bf afrag_ptr(const bf16* rowptr, int k0) {
    int kb = k0 + ((lane_() >> 4) << 4);
    return *(const v16bf*)(rowptr + kb);
}
__device__ __forceinline__ v16bf afrag(const bf16* A, int ld, int row0, int k0) {
    const bf16* rp = A + (size_t)(row0 + (lane_() & 15)) * ld;
    return afrag_ptr(rp, k0);
}
// ---- B fragment (32x16, bf16) from B^T stored row-major [n][k] ----
// lane<16: col n=lane, K {k0..k0+15}; lane>=16: K {k0+16..k0+31}
__device__ __forceinline__ v16bf bfrag(const bf16* Bt, int ld, int n0, int k0) {
    int l = lane_();
    const bf16* p = Bt + (size_t)(n0 + (l & 15)) * ld + k0 + ((l >> 4) << 4);
    return *(const v16bf*)p;
}

// ======================= cast / transpose kernels =======================
__global__ __launch_bounds__(256) void cast_x_kernel(const float* __restrict__ in,
                                                     bf16* __restrict__ out, int n) {
    int i = blockIdx.x * 256 + threadIdx.x;
    if (i < n) out[i] = (bf16)in[i];
}

// dst[n*K + k] = src[k*N + n]   (weight -> B^T bf16)
__global__ __launch_bounds__(256) void transpose_cast_kernel(const float* __restrict__ src,
                                                             bf16* __restrict__ dst,
                                                             int K, int N) {
    int i = blockIdx.x * 256 + threadIdx.x;
    if (i < K * N) {
        int k = i / N, n = i % N;
        dst[(size_t)n * K + k] = (bf16)src[i];
    }
}

// sr_w (O=512, I=512, 2, 2) -> dst[(sub*512 + o)*512 + i], sub = kh*2+kw
__global__ __launch_bounds__(256) void cast_sr_kernel(const float* __restrict__ src,
                                                      bf16* __restrict__ dst) {
    int d = blockIdx.x * 256 + threadIdx.x;           // over 4*512*512
    int sub = d >> 18, o = (d >> 9) & 511, i = d & 511;
    dst[d] = (bf16)src[(((size_t)o * 512 + i) << 2) + sub];
}

// kv (B, 256, 1024) bf16 -> vT[b][d][m], m = pk*8 + (c-512)/64
__global__ __launch_bounds__(256) void transpose_v_kernel(const bf16* __restrict__ kv,
                                                          bf16* __restrict__ vT) {
    int i = blockIdx.x * 256 + threadIdx.x;           // over 8*64*2048
    int b = i >> 17, d = (i >> 11) & 63, m = i & 2047;
    vT[i] = kv[(size_t)b * NKPOS * 2 * DIMC + (size_t)(m >> 3) * 2 * DIMC
               + DIMC + (m & 7) * HD + d];
}

// ======================= generic WMMA GEMM =======================
// C[M,N] = A[M,K](bf16,row-major) @ Bt[n][k](bf16) + bias[n]
// block: 8 waves (4 M x 2 N) -> 128x128 tile; wave: 32x64 (2x4 WMMA tiles)
__global__ __launch_bounds__(256) void gemm_bf16_kernel(
    const bf16* __restrict__ A, const bf16* __restrict__ Bt,
    const float* __restrict__ bias, float* __restrict__ outF,
    bf16* __restrict__ outB, int M, int N, int K) {
    int w = threadIdx.x >> 5;
    int wm = w >> 1, wn = w & 1;
    int m0 = blockIdx.y * 128 + wm * 32;
    int n0 = blockIdx.x * 128 + wn * 64;
    v8f c[2][4] = {};
    #pragma unroll 2
    for (int k0 = 0; k0 < K; k0 += 32) {
        v16bf a0 = afrag(A, K, m0,      k0);
        v16bf a1 = afrag(A, K, m0 + 16, k0);
        v16bf bb[4];
        #pragma unroll
        for (int j = 0; j < 4; j++) bb[j] = bfrag(Bt, K, n0 + 16 * j, k0);
        #pragma unroll
        for (int j = 0; j < 4; j++) {
            c[0][j] = wmma_bf16(a0, bb[j], c[0][j]);
            c[1][j] = wmma_bf16(a1, bb[j], c[1][j]);
        }
    }
    int l = lane_(), hi = l >> 4, nn = l & 15;
    float bs[4];
    #pragma unroll
    for (int j = 0; j < 4; j++) bs[j] = bias[n0 + 16 * j + nn];
    if (outF) {
        #pragma unroll
        for (int i = 0; i < 2; i++)
            #pragma unroll
            for (int r = 0; r < 8; r++) {
                float* rowp = outF + (size_t)(m0 + 16 * i + r + 8 * hi) * N + n0 + nn;
                #pragma unroll
                for (int j = 0; j < 4; j++) rowp[16 * j] = c[i][j][r] + bs[j];
            }
    } else {
        #pragma unroll
        for (int i = 0; i < 2; i++)
            #pragma unroll
            for (int r = 0; r < 8; r++) {
                bf16* rowp = outB + (size_t)(m0 + 16 * i + r + 8 * hi) * N + n0 + nn;
                #pragma unroll
                for (int j = 0; j < 4; j++) rowp[16 * j] = (bf16)(c[i][j][r] + bs[j]);
            }
    }
}

// ======================= conv (SR=2) as im2col GEMM =======================
// out[b, p(256), o(512)] = sum_{kh,kw,i} x[b, (2ph+kh)*32 + 2pw+kw, i] * W
__global__ __launch_bounds__(256) void conv_gemm_kernel(
    const bf16* __restrict__ xb, const bf16* __restrict__ Wt,
    const float* __restrict__ bias, float* __restrict__ out) {
    int b = blockIdx.z;
    int w = threadIdx.x >> 5;
    int wm = w >> 1, wn = w & 1;
    int m0 = blockIdx.y * 128 + wm * 32;
    int n0 = blockIdx.x * 128 + wn * 64;
    int l = lane_();
    v8f c[2][4] = {};
    #pragma unroll 2
    for (int k0 = 0; k0 < 4 * DIMC; k0 += 32) {
        int sub = k0 >> 9, kk = k0 & 511;
        int kh = sub >> 1, kw = sub & 1;
        v16bf a[2];
        #pragma unroll
        for (int i = 0; i < 2; i++) {
            int m = m0 + 16 * i + (l & 15);
            int srow = (2 * (m >> 4) + kh) * 32 + 2 * (m & 15) + kw;
            const bf16* rp = xb + ((size_t)b * NQ + srow) * DIMC + kk;
            a[i] = afrag_ptr(rp, 0);
        }
        const bf16* wbase = Wt + (size_t)sub * DIMC * DIMC;
        v16bf bb[4];
        #pragma unroll
        for (int j = 0; j < 4; j++) bb[j] = bfrag(wbase, DIMC, n0 + 16 * j, kk);
        #pragma unroll
        for (int j = 0; j < 4; j++) {
            c[0][j] = wmma_bf16(a[0], bb[j], c[0][j]);
            c[1][j] = wmma_bf16(a[1], bb[j], c[1][j]);
        }
    }
    int hi = l >> 4, nn = l & 15;
    float bs[4];
    #pragma unroll
    for (int j = 0; j < 4; j++) bs[j] = bias[n0 + 16 * j + nn];
    #pragma unroll
    for (int i = 0; i < 2; i++)
        #pragma unroll
        for (int r = 0; r < 8; r++) {
            float* rowp = out + ((size_t)b * NKPOS + m0 + 16 * i + r + 8 * hi) * DIMC
                          + n0 + nn;
            #pragma unroll
            for (int j = 0; j < 4; j++) rowp[16 * j] = c[i][j][r] + bs[j];
        }
}

// ======================= LayerNorm (per row of 512) -> bf16 =======================
__global__ __launch_bounds__(256) void ln_kernel(const float* __restrict__ in,
                                                 const float* __restrict__ g,
                                                 const float* __restrict__ bt,
                                                 bf16* __restrict__ out) {
    int row = blockIdx.x;
    const float* r = in + (size_t)row * DIMC;
    int t = threadIdx.x;
    float v0 = r[2 * t], v1 = r[2 * t + 1];
    float s = v0 + v1, s2 = v0 * v0 + v1 * v1;
    #pragma unroll
    for (int m = 1; m < 32; m <<= 1) { s += __shfl_xor(s, m); s2 += __shfl_xor(s2, m); }
    __shared__ float rs[8], rs2[8];
    if ((t & 31) == 0) { rs[t >> 5] = s; rs2[t >> 5] = s2; }
    __syncthreads();
    float S = 0.f, S2 = 0.f;
    #pragma unroll
    for (int i = 0; i < 8; i++) { S += rs[i]; S2 += rs2[i]; }
    float mu = S * (1.f / DIMC);
    float var = S2 * (1.f / DIMC) - mu * mu;
    float inv = rsqrtf(var + 1e-5f);
    bf16* o = out + (size_t)row * DIMC;
    o[2 * t]     = (bf16)((v0 - mu) * inv * g[2 * t]     + bt[2 * t]);
    o[2 * t + 1] = (bf16)((v1 - mu) * inv * g[2 * t + 1] + bt[2 * t + 1]);
}

// ======================= flash attention =======================
// block = 8 waves; wave h handles head h for one 16-query tile; keys in chunks of 64.
__global__ __launch_bounds__(256) void attn_kernel(
    const bf16* __restrict__ qb, const bf16* __restrict__ kvb,
    const bf16* __restrict__ vT, bf16* __restrict__ ao) {
    __shared__ bf16 plds[8][16][64];
    int b  = blockIdx.x >> 6;
    int n0 = (blockIdx.x & 63) * 16;
    int h  = threadIdx.x >> 5;
    int l = lane_(), hi = l >> 4, nn = l & 15;

    const bf16* qrow = qb + ((size_t)(b * NQ + n0 + nn)) * DIMC + h * HD;
    v16bf qa0 = afrag_ptr(qrow, 0);
    v16bf qa1 = afrag_ptr(qrow, 32);

    const bf16* kvbase = kvb + (size_t)b * NKPOS * 2 * DIMC;
    const bf16* vtbase = vT + (size_t)b * HD * MKV;

    v8f o[4] = {};
    float mrow[8], lrow[8];
    #pragma unroll
    for (int r = 0; r < 8; r++) { mrow[r] = -1e30f; lrow[r] = 0.f; }

    for (int ck = 0; ck < MKV / 64; ck++) {
        int mb = ck * 64;
        v8f s[4] = {};
        #pragma unroll
        for (int j = 0; j < 4; j++) {
            int key = mb + 16 * j + nn;
            const bf16* kp = kvbase + (size_t)(key >> 3) * (2 * DIMC) + (key & 7) * HD;
            // speculative prefetch of next chunk's K rows (key+64 -> +8 kv rows)
            __builtin_prefetch(kp + 8 * 2 * DIMC, 0, 1);
            v16bf bk0 = *(const v16bf*)(kp + ((l >> 4) << 4));
            v16bf bk1 = *(const v16bf*)(kp + 32 + ((l >> 4) << 4));
            s[j] = wmma_bf16(qa0, bk0, s[j]);
            s[j] = wmma_bf16(qa1, bk1, s[j]);
        }
        const float scale = 0.125f;   // hd^-0.5
        float cm[8];
        #pragma unroll
        for (int r = 0; r < 8; r++) {
            float a0 = s[0][r] * scale, a1 = s[1][r] * scale;
            float a2 = s[2][r] * scale, a3 = s[3][r] * scale;
            s[0][r] = a0; s[1][r] = a1; s[2][r] = a2; s[3][r] = a3;
            cm[r] = fmaxf(fmaxf(a0, a1), fmaxf(a2, a3));
        }
        #pragma unroll
        for (int r = 0; r < 8; r++) {
            cm[r] = fmaxf(cm[r], __shfl_xor(cm[r], 1));
            cm[r] = fmaxf(cm[r], __shfl_xor(cm[r], 2));
            cm[r] = fmaxf(cm[r], __shfl_xor(cm[r], 4));
            cm[r] = fmaxf(cm[r], __shfl_xor(cm[r], 8));
        }
        float alpha[8], rsum[8];
        #pragma unroll
        for (int r = 0; r < 8; r++) {
            float nm = fmaxf(mrow[r], cm[r]);
            alpha[r] = __expf(mrow[r] - nm);
            mrow[r] = nm;
            float p0 = __expf(s[0][r] - nm), p1 = __expf(s[1][r] - nm);
            float p2 = __expf(s[2][r] - nm), p3 = __expf(s[3][r] - nm);
            s[0][r] = p0; s[1][r] = p1; s[2][r] = p2; s[3][r] = p3;
            rsum[r] = p0 + p1 + p2 + p3;
        }
        #pragma unroll
        for (int r = 0; r < 8; r++) {
            rsum[r] += __shfl_xor(rsum[r], 1);
            rsum[r] += __shfl_xor(rsum[r], 2);
            rsum[r] += __shfl_xor(rsum[r], 4);
            rsum[r] += __shfl_xor(rsum[r], 8);
            lrow[r] = lrow[r] * alpha[r] + rsum[r];
        }
        #pragma unroll
        for (int t = 0; t < 4; t++)
            #pragma unroll
            for (int r = 0; r < 8; r++)
                o[t][r] *= alpha[r];
        // P (16x64) -> LDS in A-loadable row-major layout
        #pragma unroll
        for (int j = 0; j < 4; j++)
            #pragma unroll
            for (int r = 0; r < 8; r++)
                plds[h][r + 8 * hi][16 * j + nn] = (bf16)s[j][r];
        __syncthreads();
        const bf16* prow = &plds[h][nn][0];
        v16bf pa0 = afrag_ptr(prow, 0);
        v16bf pa1 = afrag_ptr(prow, 32);
        #pragma unroll
        for (int t = 0; t < 4; t++) {
            const bf16* vp = vtbase + (size_t)(16 * t + nn) * MKV + mb;
            // speculative prefetch of next chunk's V^T columns
            __builtin_prefetch(vp + 64, 0, 1);
            v16bf bv0 = *(const v16bf*)(vp + ((l >> 4) << 4));
            v16bf bv1 = *(const v16bf*)(vp + 32 + ((l >> 4) << 4));
            o[t] = wmma_bf16(pa0, bv0, o[t]);
            o[t] = wmma_bf16(pa1, bv1, o[t]);
        }
        __syncthreads();
    }
    float inv[8];
    #pragma unroll
    for (int r = 0; r < 8; r++) inv[r] = 1.f / lrow[r];
    #pragma unroll
    for (int t = 0; t < 4; t++)
        #pragma unroll
        for (int r = 0; r < 8; r++)
            ao[((size_t)(b * NQ + n0 + r + 8 * hi)) * DIMC + h * HD + 16 * t + nn] =
                (bf16)(o[t][r] * inv[r]);
}

// ======================= launch =======================
extern "C" void kernel_launch(void* const* d_in, const int* in_sizes, int n_in,
                              void* d_out, int out_size, void* d_ws, size_t ws_size,
                              hipStream_t stream) {
    const float* x      = (const float*)d_in[0];
    const float* q_w    = (const float*)d_in[3];
    const float* q_b    = (const float*)d_in[4];
    const float* kv_w   = (const float*)d_in[5];
    const float* kv_b   = (const float*)d_in[6];
    const float* sr_w   = (const float*)d_in[7];
    const float* sr_b   = (const float*)d_in[8];
    const float* ln_g   = (const float*)d_in[9];
    const float* ln_b   = (const float*)d_in[10];
    const float* proj_w = (const float*)d_in[11];
    const float* proj_b = (const float*)d_in[12];
    float* out = (float*)d_out;

    char* ws = (char*)d_ws;
    size_t off = 0;
    bf16* xb      = (bf16*)(ws + off); off += (size_t)BATCH * NQ * DIMC * 2;      // 8 MB
    bf16* qbuf    = (bf16*)(ws + off); off += (size_t)BATCH * NQ * DIMC * 2;      // 8 MB
    bf16* wq_t    = (bf16*)(ws + off); off += (size_t)DIMC * DIMC * 2;
    bf16* wkv_t   = (bf16*)(ws + off); off += (size_t)DIMC * 2 * DIMC * 2;
    bf16* wproj_t = (bf16*)(ws + off); off += (size_t)DIMC * DIMC * 2;
    bf16* wsr_t   = (bf16*)(ws + off); off += (size_t)4 * DIMC * DIMC * 2;
    float* convf  = (float*)(ws + off); off += (size_t)BATCH * NKPOS * DIMC * 4;  // 4 MB
    bf16* xkvn    = (bf16*)(ws + off); off += (size_t)BATCH * NKPOS * DIMC * 2;
    bf16* kvbuf   = (bf16*)(ws + off); off += (size_t)BATCH * NKPOS * 2 * DIMC * 2;
    bf16* vT      = (bf16*)(ws + off); off += (size_t)BATCH * HD * MKV * 2;
    bf16* ao      = (bf16*)(ws + off); off += (size_t)BATCH * NQ * DIMC * 2;      // 8 MB

    // 1) precision conversion / weight re-layout
    cast_x_kernel<<<(BATCH * NQ * DIMC) / 256, 256, 0, stream>>>(x, xb, BATCH * NQ * DIMC);
    transpose_cast_kernel<<<(DIMC * DIMC) / 256, 256, 0, stream>>>(q_w, wq_t, DIMC, DIMC);
    transpose_cast_kernel<<<(DIMC * 2 * DIMC) / 256, 256, 0, stream>>>(kv_w, wkv_t, DIMC, 2 * DIMC);
    transpose_cast_kernel<<<(DIMC * DIMC) / 256, 256, 0, stream>>>(proj_w, wproj_t, DIMC, DIMC);
    cast_sr_kernel<<<(4 * DIMC * DIMC) / 256, 256, 0, stream>>>(sr_w, wsr_t);

    // 2) q = x @ q_w + q_b     (M=8192, N=512, K=512) -> bf16
    gemm_bf16_kernel<<<dim3(512 / 128, (BATCH * NQ) / 128), 256, 0, stream>>>(
        xb, wq_t, q_b, nullptr, qbuf, BATCH * NQ, DIMC, DIMC);

    // 3) conv (stride-2 2x2) + bias  -> f32 (B,256,512)
    conv_gemm_kernel<<<dim3(512 / 128, NKPOS / 128, BATCH), 256, 0, stream>>>(
        xb, wsr_t, sr_b, convf);

    // 4) LayerNorm -> bf16
    ln_kernel<<<BATCH * NKPOS, 256, 0, stream>>>(convf, ln_g, ln_b, xkvn);

    // 5) kv = ln(x) @ kv_w + kv_b   (M=2048, N=1024, K=512) -> bf16
    gemm_bf16_kernel<<<dim3(1024 / 128, (BATCH * NKPOS) / 128), 256, 0, stream>>>(
        xkvn, wkv_t, kv_b, nullptr, kvbuf, BATCH * NKPOS, 2 * DIMC, DIMC);

    // 6) V^T for contiguous B-fragments in PV GEMM
    transpose_v_kernel<<<(BATCH * HD * MKV) / 256, 256, 0, stream>>>(kvbuf, vT);

    // 7) flash attention -> bf16 (B, N, 512)
    attn_kernel<<<BATCH * (NQ / 16), 256, 0, stream>>>(qbuf, kvbuf, vT, ao);

    // 8) out = attn_out @ proj_w + proj_b -> f32 d_out
    gemm_bf16_kernel<<<dim3(512 / 128, (BATCH * NQ) / 128), 256, 0, stream>>>(
        ao, wproj_t, proj_b, out, nullptr, BATCH * NQ, DIMC, DIMC);
}
